// mHCPairTransformLayer_13168369729883
// MI455X (gfx1250) — matrix operand order repaired and checked by reading, not compiled, not run on hardware
//
#include <hip/hip_runtime.h>

typedef __attribute__((ext_vector_type(16))) __bf16 v16bf;
typedef __attribute__((ext_vector_type(8)))  float  v8f;

#define TPB 256
#define MB  64            // pixels per block
#define CD  128           // channel dim C
#define HID 512           // transition hidden dim TN*C
#define APAD 8            // bf16 padding to break LDS bank conflicts

union FragBF {
  v16bf  v;
  __bf16 e[16];
  uint4  u[2];
};

__device__ __forceinline__ float sigmoidf(float x) {
  return 1.0f / (1.0f + __expf(-x));
}

// one-shot f32 -> bf16 weight conversion into workspace (W1 then W2)
__global__ __launch_bounds__(TPB, 1)
void convert_weights(const float* __restrict__ W1,
                     const float* __restrict__ W2,
                     uint2* __restrict__ dst) {
  int q = blockIdx.x * blockDim.x + threadIdx.x;   // quad index, 32768 total
  int e = q * 4;
  const float* src = (e < HID * CD) ? (W1 + e) : (W2 + (e - HID * CD));
  float4 v = *reinterpret_cast<const float4*>(src);
  union { __bf16 b[4]; uint2 u; } pk;
  pk.b[0] = (__bf16)v.x; pk.b[1] = (__bf16)v.y;
  pk.b[2] = (__bf16)v.z; pk.b[3] = (__bf16)v.w;
  dst[q] = pk.u;
}

template <bool WBF>
__global__ __launch_bounds__(TPB, 1)
void mhc_pair_fused(const float* __restrict__ p,
                    const float* __restrict__ p_mask,
                    const float* __restrict__ W_pre,
                    const float* __restrict__ W_post,
                    const float* __restrict__ W_res,
                    const float* __restrict__ b_pre,
                    const float* __restrict__ b_post,
                    const float* __restrict__ b_res,
                    const float* __restrict__ alpha_pre_p,
                    const float* __restrict__ alpha_post_p,
                    const float* __restrict__ alpha_res_p,
                    const float* __restrict__ ln_g,
                    const float* __restrict__ ln_b,
                    const float* __restrict__ W1,
                    const float* __restrict__ b1,
                    const float* __restrict__ W2,
                    const float* __restrict__ b2,
                    const __bf16* __restrict__ W1b,
                    const __bf16* __restrict__ W2b,
                    float* __restrict__ out)
{
  __shared__ float  sP[MB][CD + 1];        // input tile (f32), padded
  __shared__ float  sWeff[24][CD];         // folded routing weights
  __shared__ __bf16 sA[MB][CD + APAD];     // LayerNorm output (GEMM1 A)
  __shared__ __bf16 sH[MB][HID + APAD];    // ReLU hidden (GEMM2 A)
  __shared__ float  sSpre[MB];
  __shared__ float  sHpost[MB][4];
  __shared__ float  sRres[MB][4];

  const int tid  = threadIdx.x;
  const int pix0 = blockIdx.x * MB;

  // ---- Phase 0: pull weight streams toward L2 (global_prefetch_b8) ----
  if (WBF) {
    const char* wb = (const char*)W1b;     // W1b/W2b contiguous, 256 KB
    for (int off = tid; off < 2048; off += TPB)
      __builtin_prefetch(wb + (size_t)off * 128, 0, 1);
  } else {
    for (int off = tid; off < 2048; off += TPB) {
      __builtin_prefetch((const char*)W1 + (size_t)off * 128, 0, 1);
      __builtin_prefetch((const char*)W2 + (size_t)off * 128, 0, 1);
    }
  }

  // ---- Phase A: fold routing weights over the 4 tiled input copies ----
  for (int idx = tid; idx < 24 * CD; idx += TPB) {
    int row = idx >> 7, c = idx & (CD - 1);
    const float* src;
    if (row < 4)      src = W_pre  + row * 512;
    else if (row < 8) src = W_post + (row - 4) * 512;
    else              src = W_res  + (row - 8) * 512;
    sWeff[row][c] = src[c] + src[c + 128] + src[c + 256] + src[c + 384];
  }
  // ---- Phase B: stage input tile ----
  for (int idx = tid; idx < MB * CD; idx += TPB) {
    sP[idx >> 7][idx & (CD - 1)] = p[(size_t)pix0 * CD + idx];
  }
  __syncthreads();

  const float a_pre  = alpha_pre_p[0];
  const float a_post = alpha_post_p[0];
  const float a_res  = alpha_res_p[0];

  // ---- Phase C/D: routing scalars + Sinkhorn + LayerNorm (4 thr/pixel) ----
  {
    const int pix  = tid >> 2;
    const int part = tid & 3;
    const int cb   = part * 32;
    float pr[32];
    float ss = 0.f, sm = 0.f;
#pragma unroll
    for (int i = 0; i < 32; ++i) {
      pr[i] = sP[pix][cb + i];
      ss += pr[i] * pr[i];
      sm += pr[i];
    }
    float d[24];
#pragma unroll
    for (int r = 0; r < 24; ++r) {
      float a = 0.f;
#pragma unroll
      for (int i = 0; i < 32; ++i) a += pr[i] * sWeff[r][cb + i];
      d[r] = a;
    }
    // butterfly reduction across the 4 lanes of this pixel (wave32)
#pragma unroll
    for (int m = 1; m < 4; m <<= 1) {
      ss += __shfl_xor(ss, m, 32);
      sm += __shfl_xor(sm, m, 32);
#pragma unroll
      for (int r = 0; r < 24; ++r) d[r] += __shfl_xor(d[r], m, 32);
    }
    const float msq = ss * (1.f / CD);
    const float mu  = sm * (1.f / CD);
    const float rms = rsqrtf(msq + 1.1920929e-07f);

    float s_pre = 0.f;
#pragma unroll
    for (int n = 0; n < 4; ++n)
      s_pre += sigmoidf(a_pre * tanhf(rms * d[n]) + b_pre[n]);

    if (part == 0) {
      sSpre[pix] = s_pre;
#pragma unroll
      for (int n = 0; n < 4; ++n)
        sHpost[pix][n] = 2.f * sigmoidf(a_post * tanhf(rms * d[4 + n]) + b_post[n]);
      // Sinkhorn-Knopp on 4x4, 20 iterations
      float M4[16];
#pragma unroll
      for (int j = 0; j < 16; ++j)
        M4[j] = __expf(a_res * tanhf(rms * d[8 + j]) + b_res[j]);
      for (int it = 0; it < 20; ++it) {
#pragma unroll
        for (int m = 0; m < 4; ++m) {
          float s   = M4[4*m] + M4[4*m+1] + M4[4*m+2] + M4[4*m+3];
          float inv = __builtin_amdgcn_rcpf(s);
          M4[4*m] *= inv; M4[4*m+1] *= inv; M4[4*m+2] *= inv; M4[4*m+3] *= inv;
        }
#pragma unroll
        for (int n = 0; n < 4; ++n) {
          float s   = M4[n] + M4[n+4] + M4[n+8] + M4[n+12];
          float inv = __builtin_amdgcn_rcpf(s);
          M4[n] *= inv; M4[n+4] *= inv; M4[n+8] *= inv; M4[n+12] *= inv;
        }
      }
#pragma unroll
      for (int m = 0; m < 4; ++m)
        sRres[pix][m] = M4[4*m] + M4[4*m+1] + M4[4*m+2] + M4[4*m+3];
    }

    // LayerNorm of p_c = s_pre * p -> bf16 A operand
    const float mu_c  = s_pre * mu;
    const float var_c = s_pre * s_pre * (msq - mu * mu);
    const float ivar  = rsqrtf(var_c + 1e-5f);
#pragma unroll
    for (int i = 0; i < 32; ++i) {
      int c = cb + i;
      float mv = (s_pre * pr[i] - mu_c) * ivar * ln_g[c] + ln_b[c];
      sA[pix][c] = (__bf16)mv;
    }
  }
  __syncthreads();

  // ---- WMMA fragment loaders ----
  const int  wave = tid >> 5;
  const int  lane = tid & 31;
  const int  ln16 = lane & 15;
  const int  hi8  = (lane >> 4) * 8;   // C/D row offset for lanes 16..31
  const bool hiA  = lane >= 16;

  // A (16x32 bf16): lane<16 holds K {k0..k0+7, k0+16..k0+23}; lane>=16 +8
  auto loadA = [&](const __bf16* base, int rs, int mrow, int k0) -> v16bf {
    FragBF f;
    const __bf16* src = base + (mrow + ln16) * rs + k0 + (hiA ? 8 : 0);
    f.u[0] = *reinterpret_cast<const uint4*>(src);
    f.u[1] = *reinterpret_cast<const uint4*>(src + 16);
    return f.v;
  };
  // B (32x16): lane<16 holds K {k0..k0+7, k0+16..k0+23}; lane>=16 +8, row n of
  // row-major [N,K] weight (B[k,n] == W[n,k]); bf16 direct loads
  auto loadBb = [&](const __bf16* W, int rs, int nrow, int k0) -> v16bf {
    FragBF f;
    const __bf16* src = W + (size_t)(nrow + ln16) * rs + k0 + (hiA ? 8 : 0);
    f.u[0] = *reinterpret_cast<const uint4*>(src);
    f.u[1] = *reinterpret_cast<const uint4*>(src + 16);
    return f.v;
  };
  // fallback: f32 weight load + inline bf16 convert
  auto loadBf = [&](const float* W, int rs, int nrow, int k0) -> v16bf {
    const float* src = W + (size_t)(nrow + ln16) * rs + k0 + (hiA ? 8 : 0);
    const float4* s4 = reinterpret_cast<const float4*>(src);
    float4 f0 = s4[0], f1 = s4[1], f2 = s4[4], f3 = s4[5];
    FragBF f;
    f.e[0]=(__bf16)f0.x;  f.e[1]=(__bf16)f0.y;  f.e[2]=(__bf16)f0.z;  f.e[3]=(__bf16)f0.w;
    f.e[4]=(__bf16)f1.x;  f.e[5]=(__bf16)f1.y;  f.e[6]=(__bf16)f1.z;  f.e[7]=(__bf16)f1.w;
    f.e[8]=(__bf16)f2.x;  f.e[9]=(__bf16)f2.y;  f.e[10]=(__bf16)f2.z; f.e[11]=(__bf16)f2.w;
    f.e[12]=(__bf16)f3.x; f.e[13]=(__bf16)f3.y; f.e[14]=(__bf16)f3.z; f.e[15]=(__bf16)f3.w;
    return f.v;
  };

  const __bf16* sAp = &sA[0][0];
  const __bf16* sHp = &sH[0][0];

  // ---- GEMM1: h = relu(m @ W1^T + b1), [64,128]x[128,512] ----
  for (int nt = 0; nt < 4; ++nt) {
    const int n0 = (wave * 4 + nt) * 16;
    v8f acc0 = {}, acc1 = {}, acc2 = {}, acc3 = {};
#pragma unroll
    for (int k0 = 0; k0 < CD; k0 += 32) {
      v16bf bf = WBF ? loadBb(W1b, CD, n0, k0) : loadBf(W1, CD, n0, k0);
      acc0 = __builtin_amdgcn_wmma_f32_16x16x32_bf16(false, loadA(sAp, CD+APAD, 0,  k0),
                                                     false, bf, (short)0, acc0, false, false);
      acc1 = __builtin_amdgcn_wmma_f32_16x16x32_bf16(false, loadA(sAp, CD+APAD, 16, k0),
                                                     false, bf, (short)0, acc1, false, false);
      acc2 = __builtin_amdgcn_wmma_f32_16x16x32_bf16(false, loadA(sAp, CD+APAD, 32, k0),
                                                     false, bf, (short)0, acc2, false, false);
      acc3 = __builtin_amdgcn_wmma_f32_16x16x32_bf16(false, loadA(sAp, CD+APAD, 48, k0),
                                                     false, bf, (short)0, acc3, false, false);
    }
    const float bias = b1[n0 + ln16];
#pragma unroll
    for (int r = 0; r < 8; ++r) {
      int mrow = r + hi8;
      sH[mrow][n0 + ln16]      = (__bf16)fmaxf(acc0[r] + bias, 0.f);
      sH[mrow + 16][n0 + ln16] = (__bf16)fmaxf(acc1[r] + bias, 0.f);
      sH[mrow + 32][n0 + ln16] = (__bf16)fmaxf(acc2[r] + bias, 0.f);
      sH[mrow + 48][n0 + ln16] = (__bf16)fmaxf(acc3[r] + bias, 0.f);
    }
  }
  __syncthreads();

  // ---- GEMM2: p_t = h @ W2^T + b2, [64,512]x[512,128] ----
  const int n0 = wave * 16;
  v8f c0 = {}, c1 = {}, c2 = {}, c3 = {};
#pragma unroll
  for (int k0 = 0; k0 < HID; k0 += 32) {
    v16bf bf = WBF ? loadBb(W2b, HID, n0, k0) : loadBf(W2, HID, n0, k0);
    c0 = __builtin_amdgcn_wmma_f32_16x16x32_bf16(false, loadA(sHp, HID+APAD, 0,  k0),
                                                 false, bf, (short)0, c0, false, false);
    c1 = __builtin_amdgcn_wmma_f32_16x16x32_bf16(false, loadA(sHp, HID+APAD, 16, k0),
                                                 false, bf, (short)0, c1, false, false);
    c2 = __builtin_amdgcn_wmma_f32_16x16x32_bf16(false, loadA(sHp, HID+APAD, 32, k0),
                                                 false, bf, (short)0, c2, false, false);
    c3 = __builtin_amdgcn_wmma_f32_16x16x32_bf16(false, loadA(sHp, HID+APAD, 48, k0),
                                                 false, bf, (short)0, c3, false, false);
  }

  // ---- Epilogue: out[n] = rowsum(H_res)[n]*p + H_post[n]*(p_c + p_t*mask) ----
  const int   cc  = n0 + ln16;
  const float b2v = b2[cc];
  v8f accs[4] = {c0, c1, c2, c3};
#pragma unroll
  for (int t = 0; t < 4; ++t) {
#pragma unroll
    for (int r = 0; r < 8; ++r) {
      int   pix  = t * 16 + r + hi8;
      float pv   = sP[pix][cc];
      float pc   = sSpre[pix] * pv;
      float mask = p_mask[pix0 + pix];
      float pt   = (accs[t][r] + b2v) * mask;
      float pu   = pc + pt;
      size_t base = ((size_t)(pix0 + pix) * 4) * CD + cc;
#pragma unroll
      for (int n = 0; n < 4; ++n) {
        out[base + (size_t)n * CD] = sRres[pix][n] * pv + sHpost[pix][n] * pu;
      }
    }
  }
}

extern "C" void kernel_launch(void* const* d_in, const int* in_sizes, int n_in,
                              void* d_out, int out_size, void* d_ws, size_t ws_size,
                              hipStream_t stream) {
  (void)n_in; (void)out_size;
  const int pixels = in_sizes[0] / CD;          // B*L*L = 65536
  dim3 grid(pixels / MB);                       // 1024 blocks

  const __bf16* W1b = (const __bf16*)d_ws;
  const __bf16* W2b = W1b + HID * CD;
  const bool use_bf = ws_size >= (size_t)(2 * HID * CD) * sizeof(unsigned short);

  if (use_bf) {
    // 131072 weights, 4 per thread
    convert_weights<<<dim3((2 * HID * CD) / 4 / TPB), TPB, 0, stream>>>(
        (const float*)d_in[13], (const float*)d_in[15], (uint2*)d_ws);
    mhc_pair_fused<true><<<grid, TPB, 0, stream>>>(
        (const float*)d_in[0],  (const float*)d_in[1],  (const float*)d_in[2],
        (const float*)d_in[3],  (const float*)d_in[4],  (const float*)d_in[5],
        (const float*)d_in[6],  (const float*)d_in[7],  (const float*)d_in[8],
        (const float*)d_in[9],  (const float*)d_in[10], (const float*)d_in[11],
        (const float*)d_in[12], (const float*)d_in[13], (const float*)d_in[14],
        (const float*)d_in[15], (const float*)d_in[16], W1b, W2b, (float*)d_out);
  } else {
    mhc_pair_fused<false><<<grid, TPB, 0, stream>>>(
        (const float*)d_in[0],  (const float*)d_in[1],  (const float*)d_in[2],
        (const float*)d_in[3],  (const float*)d_in[4],  (const float*)d_in[5],
        (const float*)d_in[6],  (const float*)d_in[7],  (const float*)d_in[8],
        (const float*)d_in[9],  (const float*)d_in[10], (const float*)d_in[11],
        (const float*)d_in[12], (const float*)d_in[13], (const float*)d_in[14],
        (const float*)d_in[15], (const float*)d_in[16], W1b, W2b, (float*)d_out);
  }
}
